// CausalSelfAttention_16552803958929
// MI455X (gfx1250) — matrix operand order repaired and checked by reading, not compiled, not run on hardware
//
#include <hip/hip_runtime.h>
#include <hip/hip_bf16.h>

typedef __attribute__((ext_vector_type(16))) _Float16 v16h;
typedef __attribute__((ext_vector_type(8)))  _Float16 v8h;
typedef __attribute__((ext_vector_type(8)))  float    v8f;

#define N_HEADS 16
#define T_SEQ   2048
#define C_DIM   1024
#define D_HEAD  64

// ---------- WMMA helpers (layouts per CDNA5 ISA §7.12.2) ----------

// A fragment 16x32 f16: lane = 16*hl + m ; element e -> k = k0 + 8*hl + (e<8 ? e : e+8)
__device__ __forceinline__ v16h load_a_frag(const _Float16* __restrict__ base,
                                            int lda, int row0, int k0) {
  int lane = threadIdx.x & 31;
  int hl = lane >> 4;
  const _Float16* p = base + (size_t)(row0 + (lane & 15)) * lda + (k0 + hl * 8);
  v8h lo = *(const v8h*)(p);
  v8h hi = *(const v8h*)(p + 16);
  v16h a;
#pragma unroll
  for (int i = 0; i < 8; ++i) { a[i] = lo[i]; a[8 + i] = hi[i]; }
  return a;
}

// B fragment 32x16: B[k,n] = W[n0+(lane&15), k0 + 8*hl + (e<8 ? e : e+8)] (W row-major [N x K])
__device__ __forceinline__ v16h load_b_frag(const _Float16* __restrict__ Wbase,
                                            int ldw, int n0, int k0) {
  int lane = threadIdx.x & 31;
  int hl = lane >> 4;
  const _Float16* p = Wbase + (size_t)(n0 + (lane & 15)) * ldw + (k0 + hl * 8);
  v8h lo = *(const v8h*)(p);
  v8h hi = *(const v8h*)(p + 16);
  v16h b;
#pragma unroll
  for (int i = 0; i < 8; ++i) { b[i] = lo[i]; b[8 + i] = hi[i]; }
  return b;
}

__device__ __forceinline__ v8f wmma_f16(v16h a, v16h b, v8f c) {
  return __builtin_amdgcn_wmma_f32_16x16x32_f16(false, a, false, b, (short)0, c,
                                                false, false);
}

// ---------------------------- fp32 -> f16 convert ----------------------------
__global__ void __launch_bounds__(256)
convert_f32_to_f16(const float* __restrict__ s, _Float16* __restrict__ d, int n) {
  int i = blockIdx.x * blockDim.x + threadIdx.x;
  int stride = gridDim.x * blockDim.x;
  for (; i < n; i += stride) d[i] = (_Float16)s[i];
}

// ------------------------------- QKV GEMM ------------------------------------
// Y[m,n] = sum_k X[m,k] * Wqkv[n,k] + b[n] ; per wave: 32m x 64n register tile
// (2 A-frags x 4 B-frags -> 8 independent WMMA accumulators per K-step).
__global__ void __launch_bounds__(256)
qkv_gemm(const _Float16* __restrict__ xs, const _Float16* __restrict__ wq,
         const float* __restrict__ bias,
         _Float16* __restrict__ q16, _Float16* __restrict__ k16,
         _Float16* __restrict__ v16t) {
  int w = threadIdx.x >> 5;
  int lane = threadIdx.x & 31;
  int hl = lane >> 4;
  int ln = lane & 15;
  int mt = blockIdx.x;               // 0..127  (M=4096, 32 rows/tile)
  int nt = blockIdx.y * 8 + w;       // 0..47   (N=3072, 64 cols/tile)
  v8f acc[2][4] = {};
#pragma unroll 2
  for (int k = 0; k < C_DIM; k += 32) {
    v16h a0 = load_a_frag(xs, C_DIM, mt * 32, k);
    v16h a1 = load_a_frag(xs, C_DIM, mt * 32 + 16, k);
#pragma unroll
    for (int j = 0; j < 4; ++j) {
      v16h b = load_b_frag(wq, C_DIM, nt * 64 + j * 16, k);
      acc[0][j] = wmma_f16(a0, b, acc[0][j]);
      acc[1][j] = wmma_f16(a1, b, acc[1][j]);
    }
  }
#pragma unroll
  for (int j = 0; j < 4; ++j) {
    int n = nt * 64 + j * 16 + ln;
    float bv = bias[n];
    int three = n >> 10;             // 0=q,1=k,2=v
    int rem = n & 1023;
    int h = rem >> 6;
    int d = rem & 63;
#pragma unroll
    for (int mi = 0; mi < 2; ++mi) {
#pragma unroll
      for (int r = 0; r < 8; ++r) {
        int m = mt * 32 + mi * 16 + r + 8 * hl;  // global token row
        int b_i = m >> 11;                        // batch
        int t = m & 2047;                         // position
        _Float16 hv = (_Float16)(acc[mi][j][r] + bv);
        if (three == 0)
          q16[(((size_t)(b_i * N_HEADS + h)) * T_SEQ + t) * D_HEAD + d] = hv;
        else if (three == 1)
          k16[(((size_t)(b_i * N_HEADS + h)) * T_SEQ + t) * D_HEAD + d] = hv;
        else
          v16t[(((size_t)(b_i * N_HEADS + h)) * D_HEAD + d) * T_SEQ + t] = hv;
      }
    }
  }
}

// --------------------------- flash attention ---------------------------------
__global__ void __launch_bounds__(256)
attn_kernel(const _Float16* __restrict__ q16, const _Float16* __restrict__ k16,
            const _Float16* __restrict__ v16t, _Float16* __restrict__ a16) {
  __shared__ _Float16 plds[8][16][32];   // per-wave P staging (8 KB)
  int w = threadIdx.x >> 5;
  int lane = threadIdx.x & 31;
  int hl = lane >> 4;
  int ln = lane & 15;
  int id = blockIdx.x * 8 + w;           // 0..4095
  int qt = id & 127;                     // query tile (16 rows)
  int bh = id >> 7;                      // 0..31 (b*16 + h)
  const _Float16* qb = q16 + (size_t)bh * T_SEQ * D_HEAD;
  const _Float16* kb = k16 + (size_t)bh * T_SEQ * D_HEAD;
  const _Float16* vb = v16t + (size_t)bh * D_HEAD * T_SEQ;

  v16h qa0 = load_a_frag(qb, D_HEAD, qt * 16, 0);
  v16h qa1 = load_a_frag(qb, D_HEAD, qt * 16, 32);

  v8f acc0 = {}, acc1 = {}, acc2 = {}, acc3 = {};
  float mrow[8], lrow[8];
#pragma unroll
  for (int r = 0; r < 8; ++r) { mrow[r] = -3.0e38f; lrow[r] = 0.f; }

  const float scale = 0.125f;            // D^-0.5 = 1/8
  int ktmax = (qt * 16 + 15) >> 5;       // last 32-key tile touching causal region
  for (int kt = 0; kt <= ktmax; ++kt) {
    // ---- scores S = Q K^T (16q x 32k) as two 16x16 WMMA tiles -------------
    v8f s0 = {}, s1 = {};
    s0 = wmma_f16(qa0, load_b_frag(kb, D_HEAD, kt * 32 + 0, 0), s0);
    s0 = wmma_f16(qa1, load_b_frag(kb, D_HEAD, kt * 32 + 0, 32), s0);
    s1 = wmma_f16(qa0, load_b_frag(kb, D_HEAD, kt * 32 + 16, 0), s1);
    s1 = wmma_f16(qa1, load_b_frag(kb, D_HEAD, kt * 32 + 16, 32), s1);

    // ---- causal mask + online softmax (row stats live in 16-lane halves) --
    float p0[8], p1[8], alpha[8];
#pragma unroll
    for (int r = 0; r < 8; ++r) {
      int qrow = qt * 16 + r + 8 * hl;
      int key0 = kt * 32 + ln;
      int key1 = key0 + 16;
      float a0 = (key0 <= qrow) ? s0[r] * scale : -3.0e38f;
      float a1 = (key1 <= qrow) ? s1[r] * scale : -3.0e38f;
      float mx = fmaxf(a0, a1);
#pragma unroll
      for (int off = 1; off < 16; off <<= 1)
        mx = fmaxf(mx, __shfl_xor(mx, off, 32));
      float mnew = fmaxf(mrow[r], mx);
      float al = __expf(mrow[r] - mnew);
      float e0 = __expf(a0 - mnew);
      float e1 = __expf(a1 - mnew);
      float rs = e0 + e1;
#pragma unroll
      for (int off = 1; off < 16; off <<= 1)
        rs += __shfl_xor(rs, off, 32);
      lrow[r] = lrow[r] * al + rs;
      mrow[r] = mnew;
      alpha[r] = al;
      p0[r] = e0;
      p1[r] = e1;
    }
#pragma unroll
    for (int r = 0; r < 8; ++r) {
      acc0[r] *= alpha[r]; acc1[r] *= alpha[r];
      acc2[r] *= alpha[r]; acc3[r] *= alpha[r];
    }

    // ---- re-layout P: C-fragment -> A-fragment via per-wave LDS -----------
#pragma unroll
    for (int r = 0; r < 8; ++r) {
      plds[w][r + 8 * hl][ln] = (_Float16)p0[r];
      plds[w][r + 8 * hl][16 + ln] = (_Float16)p1[r];
    }
    asm volatile("s_wait_dscnt 0" ::: "memory");
    v16h pf;
#pragma unroll
    for (int i = 0; i < 8; ++i) {
      pf[i]     = plds[w][ln][8 * hl + i];
      pf[8 + i] = plds[w][ln][8 * hl + 16 + i];
    }
    asm volatile("s_wait_dscnt 0" ::: "memory");

    // ---- O += P V (K-dim = 32 keys, 4 d-subtiles of 16) -------------------
    acc0 = wmma_f16(pf, load_b_frag(vb, T_SEQ, 0,  kt * 32), acc0);
    acc1 = wmma_f16(pf, load_b_frag(vb, T_SEQ, 16, kt * 32), acc1);
    acc2 = wmma_f16(pf, load_b_frag(vb, T_SEQ, 32, kt * 32), acc2);
    acc3 = wmma_f16(pf, load_b_frag(vb, T_SEQ, 48, kt * 32), acc3);
  }

  // ---- normalize + write attn output as f16 [B, T, C] ---------------------
  int b_i = bh >> 4;
  int h = bh & 15;
#pragma unroll
  for (int r = 0; r < 8; ++r) {
    int qrow = qt * 16 + r + 8 * hl;
    float inv = 1.0f / lrow[r];
    size_t rowoff = ((size_t)(b_i * T_SEQ + qrow)) * C_DIM + h * D_HEAD;
    a16[rowoff + 0 * 16 + ln] = (_Float16)(acc0[r] * inv);
    a16[rowoff + 1 * 16 + ln] = (_Float16)(acc1[r] * inv);
    a16[rowoff + 2 * 16 + ln] = (_Float16)(acc2[r] * inv);
    a16[rowoff + 3 * 16 + ln] = (_Float16)(acc3[r] * inv);
  }
}

// ---------------------------- output projection ------------------------------
// out[m,n] = sum_k A[m,k] * Wproj[n,k] + b[n] ; 32m x 64n register tile.
__global__ void __launch_bounds__(256)
proj_gemm(const _Float16* __restrict__ a16, const _Float16* __restrict__ wp,
          const float* __restrict__ bias, float* __restrict__ out) {
  int w = threadIdx.x >> 5;
  int lane = threadIdx.x & 31;
  int hl = lane >> 4;
  int ln = lane & 15;
  int mt = blockIdx.x;             // 0..127 (32 rows/tile)
  int nt = blockIdx.y * 8 + w;     // 0..15  (64 cols/tile)
  v8f acc[2][4] = {};
#pragma unroll 2
  for (int k = 0; k < C_DIM; k += 32) {
    v16h a0 = load_a_frag(a16, C_DIM, mt * 32, k);
    v16h a1 = load_a_frag(a16, C_DIM, mt * 32 + 16, k);
#pragma unroll
    for (int j = 0; j < 4; ++j) {
      v16h b = load_b_frag(wp, C_DIM, nt * 64 + j * 16, k);
      acc[0][j] = wmma_f16(a0, b, acc[0][j]);
      acc[1][j] = wmma_f16(a1, b, acc[1][j]);
    }
  }
#pragma unroll
  for (int j = 0; j < 4; ++j) {
    int n = nt * 64 + j * 16 + ln;
    float bv = bias[n];
#pragma unroll
    for (int mi = 0; mi < 2; ++mi) {
#pragma unroll
      for (int r = 0; r < 8; ++r) {
        int m = mt * 32 + mi * 16 + r + 8 * hl;
        out[(size_t)m * C_DIM + n] = acc[mi][j][r] + bv;
      }
    }
  }
}

// ------------------------------- launcher ------------------------------------
extern "C" void kernel_launch(void* const* d_in, const int* in_sizes, int n_in,
                              void* d_out, int out_size, void* d_ws, size_t ws_size,
                              hipStream_t stream) {
  const float* x  = (const float*)d_in[0];   // [2, 2048, 1024]
  const float* Wq = (const float*)d_in[1];   // [3072, 1024]
  const float* bq = (const float*)d_in[2];   // [3072]
  const float* Wp = (const float*)d_in[3];   // [1024, 1024]
  const float* bp = (const float*)d_in[4];   // [1024]
  float* out = (float*)d_out;                // [2, 2048, 1024]

  char* ws = (char*)d_ws;
  _Float16* xs   = (_Float16*)(ws + 0);          //  8 MB : x f16 [4096,1024]
  _Float16* wq16 = (_Float16*)(ws + 8388608);    //  6 MB : W_qkv f16
  _Float16* wp16 = (_Float16*)(ws + 14680064);   //  2 MB : W_proj f16
  _Float16* q16  = (_Float16*)(ws + 16777216);   //  8 MB : q [B,H,T,D]
  _Float16* k16  = (_Float16*)(ws + 25165824);   //  8 MB : k [B,H,T,D]
  _Float16* v16t = (_Float16*)(ws + 33554432);   //  8 MB : v [B,H,D,T]
  _Float16* a16  = (_Float16*)(ws + 41943040);   //  8 MB : attn out [B,T,C]

  convert_f32_to_f16<<<1024, 256, 0, stream>>>(x,  xs,   4096 * 1024);
  convert_f32_to_f16<<<1024, 256, 0, stream>>>(Wq, wq16, 3072 * 1024);
  convert_f32_to_f16<<<512,  256, 0, stream>>>(Wp, wp16, 1024 * 1024);

  qkv_gemm<<<dim3(128, 6), 256, 0, stream>>>(xs, wq16, bq, q16, k16, v16t);
  attn_kernel<<<512, 256, 0, stream>>>(q16, k16, v16t, a16);
  proj_gemm<<<dim3(128, 2), 256, 0, stream>>>(a16, wp16, bp, out);
}
